// GQAAttention_16793322127768
// MI455X (gfx1250) — matrix operand order repaired and checked by reading, not compiled, not run on hardware
//
#include <hip/hip_runtime.h>
#include <hip/hip_bf16.h>

// ---------------------------------------------------------------------------
// GQA attention with ternarized weights, CDNA5 (gfx1250) WMMA implementation.
// B=2, S=2048, DIM=2048, 32 heads, 8 KV heads, head_dim=64, causal.
// Matmuls via v_wmma_f32_16x16x32_f16 (wave32); GEMM tiles staged with
// global_load_async_to_lds_b128 + s_wait_asynccnt (ASYNCcnt-tracked).
// ---------------------------------------------------------------------------

typedef __attribute__((ext_vector_type(16))) _Float16 v16h;
typedef __attribute__((ext_vector_type(8)))  float    v8f;
typedef int v4i_ __attribute__((vector_size(16)));

#define S_LEN 2048
#define DIM   2048
#define KVDIM 512
#define HD    64
#define NHEAD 32
#define NKV   8

#if defined(__HIP_DEVICE_COMPILE__) && \
    __has_builtin(__builtin_amdgcn_global_load_async_to_lds_b128)
#define ASYNC_LDS 1
#else
#define ASYNC_LDS 0
#endif

template <int N>
__device__ __forceinline__ void wait_asynccnt_c() {
#if __has_builtin(__builtin_amdgcn_s_wait_asynccnt)
    __builtin_amdgcn_s_wait_asynccnt(N);
#elif ASYNC_LDS
    asm volatile("s_wait_asynccnt %0" ::"i"(N) : "memory");
#endif
}

#if ASYNC_LDS
__device__ __forceinline__ void async_copy_b128(const _Float16* g, _Float16* l) {
    __builtin_amdgcn_global_load_async_to_lds_b128(
        (__attribute__((address_space(1))) v4i_*)g,
        (__attribute__((address_space(3))) v4i_*)l, 0, 0);
}
#endif

// --- WMMA fragment loaders (ISA 7.12.2 layouts, wave32) ---------------------

union FragU { v16h v; unsigned u[8]; };

// A-matrix 16x32 f16: lane L (L<16) holds row L; VGPR0..3 -> K=0..7,
// VGPR4..7 -> K=16..23. Lane L+16 holds row L with K offset by 8.
__device__ __forceinline__ v16h load_a_frag(const _Float16* base, int ldm) {
    int lane = threadIdx.x & 31;
    int row  = lane & 15;
    int c0   = (lane < 16) ? 0 : 4;               // u32 offset (K offset 0 or 8)
    const unsigned* pu = (const unsigned*)(base + (size_t)row * ldm);
    FragU f;
#pragma unroll
    for (int v = 0; v < 8; ++v)
        f.u[v] = pu[c0 + ((v < 4) ? v : v + 4)];  // K = kb + {0,2,4,6,16,18,20,22}
    return f.v;
}

// B-matrix 32x16 f16 supplied as rows of an [N][K] array (B[k][n] = W[n][k]):
// lane L (L<16) holds column n=L with K=0..7,16..23; lane L+16 -> K=8..15,24..31.
__device__ __forceinline__ v16h load_b_frag(const _Float16* base, int ldn) {
    int lane = threadIdx.x & 31;
    int n    = lane & 15;
    int kb   = (lane < 16) ? 0 : 8;
    const unsigned* pu = (const unsigned*)(base + (size_t)n * ldn + kb);
    FragU f;
#pragma unroll
    for (int v = 0; v < 8; ++v) f.u[v] = pu[(v < 4) ? v : v + 4];
    return f.v;
}

__device__ __forceinline__ v8f wmma_f16(v16h a, v16h b, v8f c) {
    return __builtin_amdgcn_wmma_f32_16x16x32_f16(false, a, false, b,
                                                  (short)0, c, false, false);
}

// --- Weight ternarization ---------------------------------------------------

__global__ __launch_bounds__(256) void absmean_kernel(const float* __restrict__ w,
                                                      int n, float* __restrict__ sum) {
    __shared__ float red[8];
    int tid = threadIdx.x;
    float s = 0.f;
    for (size_t i = (size_t)blockIdx.x * 256 + tid; i < (size_t)n;
         i += (size_t)gridDim.x * 256)
        s += fabsf(w[i]);
#pragma unroll
    for (int off = 16; off > 0; off >>= 1) s += __shfl_xor(s, off, 32);
    if ((tid & 31) == 0) red[tid >> 5] = s;
    __syncthreads();
    if (tid < 8) {
        float t = red[tid];
#pragma unroll
        for (int off = 4; off > 0; off >>= 1) t += __shfl_xor(t, off, 32);
        if (tid == 0) atomicAdd(sum, t);
    }
}

__global__ __launch_bounds__(256) void ternarize_kernel(const float* __restrict__ w,
                                                        _Float16* __restrict__ o,
                                                        int n,
                                                        const float* __restrict__ sum) {
    float scale = fmaxf(*sum / (float)n, 1e-6f);
    float thr   = 0.05f * scale;
    for (size_t i = (size_t)blockIdx.x * 256 + threadIdx.x; i < (size_t)n;
         i += (size_t)gridDim.x * 256) {
        float x = w[i];
        o[i] = (_Float16)((x > thr) ? 1.f : ((x < -thr) ? -1.f : 0.f));
    }
}

__global__ __launch_bounds__(256) void f32_to_f16_kernel(const float* __restrict__ in,
                                                         _Float16* __restrict__ o, int n) {
    for (size_t i = (size_t)blockIdx.x * 256 + threadIdx.x; i < (size_t)n;
         i += (size_t)gridDim.x * 256)
        o[i] = (_Float16)in[i];
}

// --- Tiled WMMA GEMM: C[M,N] = scale * A[M,K] x B^T (B stored [N][K]) -------
// OUT_MODE: 0 = f16 row-major, 1 = f16 transposed [N][M], 2 = f32 row-major.
// M%128==0, N%64==0, K%32==0 (all shapes here satisfy this).

template <int OUT_MODE>
__global__ __launch_bounds__(256) void gemm_wmma(const _Float16* __restrict__ A,
                                                 const _Float16* __restrict__ B,
                                                 void* __restrict__ C,
                                                 int M, int N, int K, float scale) {
#if ASYNC_LDS
    __shared__ _Float16 As[2][128 * 32];
    __shared__ _Float16 Bs[2][64 * 32];
#else
    __shared__ _Float16 As[1][128 * 32];
    __shared__ _Float16 Bs[1][64 * 32];
#endif
    int m0   = blockIdx.y * 128;
    int n0   = blockIdx.x * 64;
    int tid  = threadIdx.x;
    int wave = tid >> 5;
    int lane = tid & 31;
    int wm   = (wave >> 1) * 32;
    int wn   = (wave & 1) * 32;

    v8f acc[2][2] = {};

#if ASYNC_LDS
    // Async tile staging: global -> LDS, 16B per transfer, ASYNCcnt-tracked.
    // A tile: 128x32 halves = 512 xfers (2/thread); B tile: 64x32 = 256 (1/thread).
    auto issue_stage = [&](int k0, int buf) {
#pragma unroll
        for (int i = 0; i < 2; ++i) {
            int lin = tid + i * 256;
            int r = lin >> 2, c = lin & 3;
            async_copy_b128(A + (size_t)(m0 + r) * K + k0 + c * 8,
                            &As[buf][r * 32 + c * 8]);
        }
        {
            int r = tid >> 2, c = tid & 3;
            async_copy_b128(B + (size_t)(n0 + r) * K + k0 + c * 8,
                            &Bs[buf][r * 32 + c * 8]);
        }
    };
    int nsteps = K / 32;
    issue_stage(0, 0);
    for (int s = 0; s < nsteps; ++s) {
        if (s + 1 < nsteps) {
            issue_stage((s + 1) * 32, (s + 1) & 1);
            wait_asynccnt_c<3>();                 // stage s's 3 xfers complete
        } else {
            wait_asynccnt_c<0>();
        }
        __syncthreads();
        int buf = s & 1;
        v16h a0 = load_a_frag(As[buf] + (wm +  0) * 32, 32);
        v16h a1 = load_a_frag(As[buf] + (wm + 16) * 32, 32);
        v16h b0 = load_b_frag(Bs[buf] + (wn +  0) * 32, 32);
        v16h b1 = load_b_frag(Bs[buf] + (wn + 16) * 32, 32);
        acc[0][0] = wmma_f16(a0, b0, acc[0][0]);
        acc[0][1] = wmma_f16(a0, b1, acc[0][1]);
        acc[1][0] = wmma_f16(a1, b0, acc[1][0]);
        acc[1][1] = wmma_f16(a1, b1, acc[1][1]);
        __syncthreads();
    }
#else
    const unsigned* Ag  = (const unsigned*)A;
    const unsigned* Bg  = (const unsigned*)B;
    unsigned*       AsU = (unsigned*)As[0];
    unsigned*       BsU = (unsigned*)Bs[0];
    for (int k0 = 0; k0 < K; k0 += 32) {
        __syncthreads();
#pragma unroll
        for (int i = 0; i < 8; ++i) {             // A tile 128x32 halves
            int lin = tid + i * 256;
            int r = lin >> 4, c = lin & 15;
            AsU[lin] = Ag[((size_t)(m0 + r) * K + k0) / 2 + c];
        }
#pragma unroll
        for (int i = 0; i < 4; ++i) {             // B tile 64x32 halves
            int lin = tid + i * 256;
            int r = lin >> 4, c = lin & 15;
            BsU[lin] = Bg[((size_t)(n0 + r) * K + k0) / 2 + c];
        }
        __syncthreads();
        v16h a0 = load_a_frag(As[0] + (wm +  0) * 32, 32);
        v16h a1 = load_a_frag(As[0] + (wm + 16) * 32, 32);
        v16h b0 = load_b_frag(Bs[0] + (wn +  0) * 32, 32);
        v16h b1 = load_b_frag(Bs[0] + (wn + 16) * 32, 32);
        acc[0][0] = wmma_f16(a0, b0, acc[0][0]);
        acc[0][1] = wmma_f16(a0, b1, acc[0][1]);
        acc[1][0] = wmma_f16(a1, b0, acc[1][0]);
        acc[1][1] = wmma_f16(a1, b1, acc[1][1]);
    }
#endif

    int cn   = lane & 15;
    int radd = (lane < 16) ? 0 : 8;
#pragma unroll
    for (int i = 0; i < 2; ++i)
#pragma unroll
        for (int j = 0; j < 2; ++j)
#pragma unroll
            for (int r = 0; r < 8; ++r) {
                int   row = m0 + wm + i * 16 + r + radd;
                int   col = n0 + wn + j * 16 + cn;
                float val = acc[i][j][r] * scale;
                if (OUT_MODE == 0)
                    ((_Float16*)C)[(size_t)row * N + col] = (_Float16)val;
                else if (OUT_MODE == 1)
                    ((_Float16*)C)[(size_t)col * M + row] = (_Float16)val;
                else
                    ((float*)C)[(size_t)row * N + col] = val;
            }
}

// --- Flash-style causal GQA attention --------------------------------------
// Q  : [B*S][DIM]  f16 (pre-scaled by 1/sqrt(hd)), head h occupies cols h*64..
// Kb : [B*S][KVDIM] f16, kv-group g occupies cols g*64..
// Vt : [KVDIM][B*S] f16 transposed (row = g*64+d, col = b*2048+s)
// Ctx: [B*S][DIM]  f16 output context
// grid = (S/64, NHEAD, B), block = 128 (4 waves, 16 query rows per wave)
// Row-sum of P computed on the matrix unit via an all-ones B operand
// (l-accumulator rescaled by alpha exactly like the output accumulator).

__global__ __launch_bounds__(128) void attn_kernel(const _Float16* __restrict__ Q,
                                                   const _Float16* __restrict__ Kb,
                                                   const _Float16* __restrict__ Vt,
                                                   _Float16* __restrict__ Ctx) {
    __shared__ _Float16 Ps[64 * 64];
    int qb   = blockIdx.x;
    int h    = blockIdx.y;
    int b    = blockIdx.z;
    int g    = h >> 2;
    int wave = threadIdx.x >> 5;
    int lane = threadIdx.x & 31;
    int q0   = qb * 64;
    int qrow0 = q0 + wave * 16;
    int cn   = lane & 15;
    int radd = (lane < 16) ? 0 : 8;

    const _Float16* Qbase = Q + ((size_t)(b * S_LEN + qrow0)) * DIM + h * HD;
    v16h aq0 = load_a_frag(Qbase, DIM);       // d = 0..31
    v16h aq1 = load_a_frag(Qbase + 32, DIM);  // d = 32..63

    v16h ones;
#pragma unroll
    for (int i = 0; i < 16; ++i) ones[i] = (_Float16)1.0f;

    v8f   oacc[4] = {};
    v8f   lacc    = {};
    float mrun[8];
#pragma unroll
    for (int r = 0; r < 8; ++r) mrun[r] = -1e30f;

    for (int jb = 0; jb <= qb; ++jb) {
        int j0 = jb * 64;
        v8f s[4];
#pragma unroll
        for (int nf = 0; nf < 4; ++nf) {      // 16 keys per fragment
            const _Float16* Kbase =
                Kb + ((size_t)(b * S_LEN + j0 + nf * 16)) * KVDIM + g * HD;
            v16h bk0 = load_b_frag(Kbase, KVDIM);
            v16h bk1 = load_b_frag(Kbase + 32, KVDIM);
            v8f  z   = {};
            z = wmma_f16(aq0, bk0, z);
            z = wmma_f16(aq1, bk1, z);
            s[nf] = z;
        }

        if (jb == qb) {                        // causal mask on diagonal block
#pragma unroll
            for (int nf = 0; nf < 4; ++nf)
#pragma unroll
                for (int r = 0; r < 8; ++r) {
                    int row = qrow0 + r + radd;
                    int col = j0 + nf * 16 + cn;
                    if (col > row) s[nf][r] = -1e30f;
                }
        }

        // online softmax: row-max via half-wave shuffles; row-sum via WMMA
#pragma unroll
        for (int r = 0; r < 8; ++r) {
            float mx = fmaxf(fmaxf(s[0][r], s[1][r]), fmaxf(s[2][r], s[3][r]));
#pragma unroll
            for (int off = 1; off < 16; off <<= 1) mx = fmaxf(mx, __shfl_xor(mx, off, 32));
            float mnew  = fmaxf(mrun[r], mx);
            float alpha = __expf(mrun[r] - mnew);
            mrun[r] = mnew;
#pragma unroll
            for (int nf = 0; nf < 4; ++nf) {
                s[nf][r] = __expf(s[nf][r] - mnew);
                oacc[nf][r] *= alpha;
            }
            lacc[r] *= alpha;
        }

        // C-layout P -> LDS -> A-layout P (wave-private 16x64 strip)
#pragma unroll
        for (int nf = 0; nf < 4; ++nf)
#pragma unroll
            for (int r = 0; r < 8; ++r)
                Ps[(wave * 16 + r + radd) * 64 + nf * 16 + cn] = (_Float16)s[nf][r];

        v16h ap0 = load_a_frag(Ps + wave * 16 * 64, 64);       // keys 0..31
        v16h ap1 = load_a_frag(Ps + wave * 16 * 64 + 32, 64);  // keys 32..63

        lacc = wmma_f16(ap0, ones, lacc);      // l += P . 1  (row sums)
        lacc = wmma_f16(ap1, ones, lacc);

#pragma unroll
        for (int nf = 0; nf < 4; ++nf) {       // 16 output dims per fragment
            const _Float16* Vbase =
                Vt + ((size_t)(g * HD + nf * 16)) * (2 * S_LEN) + b * S_LEN + j0;
            v16h bv0 = load_b_frag(Vbase, 2 * S_LEN);
            v16h bv1 = load_b_frag(Vbase + 32, 2 * S_LEN);
            oacc[nf] = wmma_f16(ap0, bv0, oacc[nf]);
            oacc[nf] = wmma_f16(ap1, bv1, oacc[nf]);
        }
    }

#pragma unroll
    for (int nf = 0; nf < 4; ++nf)
#pragma unroll
        for (int r = 0; r < 8; ++r) {
            float val = oacc[nf][r] / lacc[r];
            int   row = qrow0 + r + radd;
            Ctx[((size_t)(b * S_LEN + row)) * DIM + h * HD + nf * 16 + cn] =
                (_Float16)val;
        }
}

// ---------------------------------------------------------------------------

extern "C" void kernel_launch(void* const* d_in, const int* in_sizes, int n_in,
                              void* d_out, int out_size, void* d_ws, size_t ws_size,
                              hipStream_t stream) {
    const float* x  = (const float*)d_in[0];   // [2,2048,2048]
    const float* wq = (const float*)d_in[1];   // [2048,2048]
    const float* wk = (const float*)d_in[2];   // [512,2048]
    const float* wv = (const float*)d_in[3];   // [512,2048]
    const float* wo = (const float*)d_in[4];   // [2048,2048]

    const int nX  = 2 * S_LEN * DIM;           // 8388608
    const int nWq = DIM * DIM;                 // 4194304
    const int nWk = KVDIM * DIM;               // 1048576
    const int nWv = KVDIM * DIM;
    const int nWo = DIM * DIM;
    const int M   = 2 * S_LEN;                 // 4096 (flattened b,s)

    unsigned char* ws = (unsigned char*)d_ws;
    float* sums = (float*)ws;                                 // 4 accumulators
    size_t off = 256;
    _Float16* xh  = (_Float16*)(ws + off); off += (size_t)nX  * 2;
    _Float16* wqh = (_Float16*)(ws + off); off += (size_t)nWq * 2;
    _Float16* wkh = (_Float16*)(ws + off); off += (size_t)nWk * 2;
    _Float16* wvh = (_Float16*)(ws + off); off += (size_t)nWv * 2;
    _Float16* woh = (_Float16*)(ws + off); off += (size_t)nWo * 2;
    _Float16* qh  = (_Float16*)(ws + off); off += (size_t)M * DIM   * 2;
    _Float16* kh  = (_Float16*)(ws + off); off += (size_t)M * KVDIM * 2;
    _Float16* vth = (_Float16*)(ws + off); off += (size_t)M * KVDIM * 2;
    _Float16* ctx = (_Float16*)(ws + off); off += (size_t)M * DIM   * 2;

    (void)hipMemsetAsync(sums, 0, 4 * sizeof(float), stream);

    absmean_kernel<<<512, 256, 0, stream>>>(wq, nWq, sums + 0);
    absmean_kernel<<<512, 256, 0, stream>>>(wk, nWk, sums + 1);
    absmean_kernel<<<512, 256, 0, stream>>>(wv, nWv, sums + 2);
    absmean_kernel<<<512, 256, 0, stream>>>(wo, nWo, sums + 3);

    ternarize_kernel<<<2048, 256, 0, stream>>>(wq, wqh, nWq, sums + 0);
    ternarize_kernel<<<1024, 256, 0, stream>>>(wk, wkh, nWk, sums + 1);
    ternarize_kernel<<<1024, 256, 0, stream>>>(wv, wvh, nWv, sums + 2);
    ternarize_kernel<<<2048, 256, 0, stream>>>(wo, woh, nWo, sums + 3);
    f32_to_f16_kernel<<<2048, 256, 0, stream>>>(x, xh, nX);

    // Q = x @ wq^T, pre-scaled by 1/sqrt(64)
    gemm_wmma<0><<<dim3(DIM / 64, M / 128), 256, 0, stream>>>(
        xh, wqh, qh, M, DIM, DIM, 0.125f);
    // K = x @ wk^T
    gemm_wmma<0><<<dim3(KVDIM / 64, M / 128), 256, 0, stream>>>(
        xh, wkh, kh, M, KVDIM, DIM, 1.0f);
    // V^T = (x @ wv^T)^T  (transposed store for contiguous PV B-fragments)
    gemm_wmma<1><<<dim3(KVDIM / 64, M / 128), 256, 0, stream>>>(
        xh, wvh, vth, M, KVDIM, DIM, 1.0f);

    attn_kernel<<<dim3(S_LEN / 64, NHEAD, 2), 128, 0, stream>>>(qh, kh, vth, ctx);

    // out = ctx @ wo^T (f32 output)
    gemm_wmma<2><<<dim3(DIM / 64, M / 128), 256, 0, stream>>>(
        ctx, woh, d_out, M, DIM, DIM, 1.0f);
}
